// Yolact_79267916415217
// MI455X (gfx1250) — compile-verified, matrix operand order checked by reading, not run on hardware
//
#include <hip/hip_runtime.h>
#include <hip/hip_bf16.h>

typedef __attribute__((ext_vector_type(16))) _Float16 v16h;
typedef __attribute__((ext_vector_type(8)))  _Float16 v8h;
typedef __attribute__((ext_vector_type(8)))  float    v8f;

#define BM 64      // block tile: out channels
#define BN 128     // block tile: spatial positions
#define BK 32      // K step (one WMMA depth)
#define ASTR 40    // LDS row stride (f16): 80B rows -> b128 reads hit all 64 banks
#define BSTR 40

// ---------------------------------------------------------------------------
// Implicit-GEMM convolution, NHWC activations, LDS-tiled, V_WMMA_F32_16X16X32_F16.
//   GEMM view: M = Cout, N = B*OH*OW, K = KH*KW*Cin, ordered kk = rs*Cin + ci.
//   Requires: K % 32 == 0, Cin power of two (ciShift = log2(Cin)).
// Weights pre-permuted to [oc][rs*Cin+ci] f16; activations NHWC f16.
// B staging: an 8-chunk of kk at fixed rs = 8 contiguous channels of one pixel
//   -> one 16B global load (two per thread per K-step).
// Output NHWC: 8 acc rows are 8 contiguous out-channels -> one 16B store/tile.
// ---------------------------------------------------------------------------
__global__ __launch_bounds__(256) void conv_wmma_kernel(
    const _Float16* __restrict__ in, const _Float16* __restrict__ wt,
    const float* __restrict__ bias, _Float16* __restrict__ out,
    int B, int Cin, int H, int W, int Cout, int KH, int KW,
    int stride, int pad, int OH, int OW, int act, int ciShift,
    int mBlocks, int nBlocks)
{
    __shared__ _Float16 As[BM * ASTR];   // [m][k]
    __shared__ _Float16 Bs[BN * BSTR];   // [n][k]

    const int mb = blockIdx.x % mBlocks;
    const int nb = blockIdx.x / mBlocks;
    const int tid  = threadIdx.x;
    const int wid  = tid >> 5;
    const int lane = tid & 31;
    const int half = lane >> 4;
    const int r16  = lane & 15;
    const int waveM = wid & 1;           // 0..1 -> 32-row band
    const int waveN = wid >> 1;          // 0..3 -> 32-col band

    const int K    = Cin * KH * KW;
    const int Ntot = B * OH * OW;
    const int hw   = OH * OW;
    const int ocBase = mb * BM;
    const int nBase  = nb * BN;

    // ---- gather-side per-thread coordinates (constant over K loop) ----
    const int m_a  = tid >> 2;           // A row this thread stages
    const int kkb  = (tid & 3) << 3;     // 8-element k chunk
    const int ocA  = ocBase + m_a;
    const bool ocAValid = ocA < Cout;
    const _Float16* wrow = wt + (long long)(ocAValid ? ocA : 0) * K;

    int nrow[2];  nrow[0] = tid >> 2;  nrow[1] = (tid >> 2) + 64;
    bool nValidG[2]; int gb[2], giy0[2], gix0[2];
#pragma unroll
    for (int q = 0; q < 2; ++q) {
        int n = nBase + nrow[q];
        nValidG[q] = n < Ntot;
        int nc  = nValidG[q] ? n : 0;
        int bi  = nc / hw, rem = nc - bi * hw;
        int oy  = rem / OW, ox = rem - oy * OW;
        gb[q]   = bi;
        giy0[q] = oy * stride - pad;
        gix0[q] = ox * stride - pad;
    }

    // ---- store-side per-lane coordinates ----
    int sb[2], soy[2], sox[2]; bool sValid[2];
#pragma unroll
    for (int g = 0; g < 2; ++g) {
        int n = nBase + waveN * 32 + g * 16 + r16;
        sValid[g] = n < Ntot;
        int nc = sValid[g] ? n : 0;
        int bi = nc / hw, rem = nc - bi * hw;
        sb[g] = bi; soy[g] = rem / OW; sox[g] = rem - soy[g] * OW;
    }

    v8f acc[2][2] = {};

    for (int k0 = 0; k0 < K; k0 += BK) {
        __syncthreads();   // previous iteration's fragment reads complete

        // ---- stage A: one 16B global load -> one 16B LDS store ----
        {
            v8h a = {};
            if (ocAValid) a = *(const v8h*)(wrow + k0 + kkb);
            *(v8h*)&As[m_a * ASTR + kkb] = a;
        }

        // ---- stage B: one 16B NHWC load per 8-chunk (fixed rs) ----
#pragma unroll
        for (int q = 0; q < 2; ++q) {
            v8h bvec = {};
            if (nValidG[q]) {
                int kkc = k0 + kkb;
                int rs  = kkc >> ciShift;
                int ci0 = kkc - (rs << ciShift);
                int r   = rs / KW, s = rs - r * KW;
                int iy  = giy0[q] + r, ix = gix0[q] + s;
                if (iy >= 0 && iy < H && ix >= 0 && ix < W)
                    bvec = *(const v8h*)(in +
                        (((long long)gb[q] * H + iy) * W + ix) * Cin + ci0);
            }
            *(v8h*)&Bs[nrow[q] * BSTR + kkb] = bvec;
        }

        __syncthreads();

        // ---- fragments: two b128 LDS reads each; then 4 WMMAs ----
        v16h af[2], bf[2];
#pragma unroll
        for (int f = 0; f < 2; ++f) {
            int m = waveM * 32 + f * 16 + r16;
            v8h lo = *(const v8h*)&As[m * ASTR + (half << 3)];        // kk = half*8 + e
            v8h hi = *(const v8h*)&As[m * ASTR + 16 + (half << 3)];   // kk = 16 + half*8 + e
#pragma unroll
            for (int e = 0; e < 8; ++e) { af[f][e] = lo[e]; af[f][8 + e] = hi[e]; }
        }
#pragma unroll
        for (int g = 0; g < 2; ++g) {
            int nl = waveN * 32 + g * 16 + r16;
            v8h lo = *(const v8h*)&Bs[nl * BSTR + (half << 3)];
            v8h hi = *(const v8h*)&Bs[nl * BSTR + 16 + (half << 3)];
#pragma unroll
            for (int e = 0; e < 8; ++e) { bf[g][e] = lo[e]; bf[g][8 + e] = hi[e]; }
        }
#pragma unroll
        for (int f = 0; f < 2; ++f)
#pragma unroll
            for (int g = 0; g < 2; ++g)
                acc[f][g] = __builtin_amdgcn_wmma_f32_16x16x32_f16(
                    false, af[f], false, bf[g], (short)0, acc[f][g], false, false);
    }

    // ---- store NHWC: acc rows are contiguous out-channels ----
    const bool vecOK = (Cout & 7) == 0;
#pragma unroll
    for (int f = 0; f < 2; ++f) {
#pragma unroll
        for (int g = 0; g < 2; ++g) {
            if (!sValid[g]) continue;
            int oc0 = ocBase + waveM * 32 + f * 16 + (half << 3);
            long long base = (((long long)sb[g] * OH + soy[g]) * OW + sox[g]) * Cout;
            if (vecOK && oc0 + 8 <= Cout) {
                v8h o;
#pragma unroll
                for (int r = 0; r < 8; ++r) {
                    float v = acc[f][g][r] + bias[oc0 + r];
                    if (act == 1) v = fmaxf(v, 0.0f);
                    o[r] = (_Float16)v;
                }
                *(v8h*)(out + base + oc0) = o;
            } else {
#pragma unroll
                for (int r = 0; r < 8; ++r) {
                    if (oc0 + r < Cout) {
                        float v = acc[f][g][r] + bias[oc0 + r];
                        if (act == 1) v = fmaxf(v, 0.0f);
                        out[base + oc0 + r] = (_Float16)v;
                    }
                }
            }
        }
    }
}

// ---------------------------------------------------------------------------
// Weight re-layout: OIHW f32 -> f16 [oc][rs*Cin + ci]
// ---------------------------------------------------------------------------
__global__ void weight_rsc_f16_kernel(const float* __restrict__ src,
                                      _Float16* __restrict__ dst,
                                      int Cout, int Cin, int KHW)
{
    long long idx = (long long)blockIdx.x * blockDim.x + threadIdx.x;
    long long total = (long long)Cout * Cin * KHW;
    if (idx >= total) return;
    int ci = (int)(idx % Cin);
    long long t = idx / Cin;
    int rs = (int)(t % KHW);
    int oc = (int)(t / KHW);
    dst[idx] = (_Float16)src[((long long)oc * Cin + ci) * KHW + rs];
}

// NCHW f32 -> NHWC f16 (inputs)
__global__ void nchw_to_nhwc_f16_kernel(const float* __restrict__ src,
                                        _Float16* __restrict__ dst,
                                        int B, int C, int H, int W)
{
    long long idx = (long long)blockIdx.x * blockDim.x + threadIdx.x;
    long long total = (long long)B * C * H * W;
    if (idx >= total) return;
    int c = (int)(idx % C);
    long long t = idx / C;
    int x = (int)(t % W); t /= W;
    int y = (int)(t % H);
    int b = (int)(t / H);
    dst[idx] = (_Float16)src[(((long long)b * C + c) * H + y) * W + x];
}

// f16 -> f32 copy (proto output is already NHWC)
__global__ void f16_to_f32_kernel(const _Float16* __restrict__ src,
                                  float* __restrict__ dst, long long n)
{
    long long i = (long long)blockIdx.x * blockDim.x + threadIdx.x;
    if (i < n) dst[i] = (float)src[i];
}

// ---------------------------------------------------------------------------
// Half-pixel bilinear resize, NHWC f16 [B,Hs,Ws,C] -> [B,Hd,Wd,C], opt. accum.
// Channel-innermost indexing -> fully coalesced.
// ---------------------------------------------------------------------------
__global__ void bilinear_kernel(const _Float16* __restrict__ src, _Float16* __restrict__ dst,
                                int B, int C, int Hs, int Ws, int Hd, int Wd, int accumulate)
{
    long long idx = (long long)blockIdx.x * blockDim.x + threadIdx.x;
    long long total = (long long)B * Hd * Wd * C;
    if (idx >= total) return;
    int c = (int)(idx % C);
    long long t = idx / C;
    int x = (int)(t % Wd); t /= Wd;
    int y = (int)(t % Hd);
    int b = (int)(t / Hd);

    float fy = (y + 0.5f) * (float)Hs / (float)Hd - 0.5f;
    float fx = (x + 0.5f) * (float)Ws / (float)Wd - 0.5f;
    int y0 = (int)floorf(fy), x0 = (int)floorf(fx);
    float wy = fy - (float)y0, wx = fx - (float)x0;
    int y0c = min(max(y0, 0), Hs - 1), y1c = min(max(y0 + 1, 0), Hs - 1);
    int x0c = min(max(x0, 0), Ws - 1), x1c = min(max(x0 + 1, 0), Ws - 1);

    const _Float16* p = src + (long long)b * Hs * Ws * C;
    float v = (1.0f - wy) * ((1.0f - wx) * (float)p[((long long)y0c * Ws + x0c) * C + c]
                          +          wx  * (float)p[((long long)y0c * Ws + x1c) * C + c])
            +          wy * ((1.0f - wx) * (float)p[((long long)y1c * Ws + x0c) * C + c]
                          +          wx  * (float)p[((long long)y1c * Ws + x1c) * C + c]);
    if (accumulate) v += (float)dst[idx];
    dst[idx] = (_Float16)v;
}

// ---------------------------------------------------------------------------
// Head output: src NHWC f16 [B,H,W,3*D] ->
//   dst[(b*Ptot + priorOff + (y*W+x)*3 + a)*D + d] (f32), c = a*D + d
// ---------------------------------------------------------------------------
__global__ void head_permute_kernel(const _Float16* __restrict__ src, float* __restrict__ dst,
                                    int B, int H, int W, int D, int priorOff, int Ptot,
                                    int applyTanh)
{
    long long idx = (long long)blockIdx.x * blockDim.x + threadIdx.x;
    long long total = (long long)B * H * W * 3 * D;
    if (idx >= total) return;
    int c = (int)(idx % (3 * D));
    long long t = idx / (3 * D);
    int x = (int)(t % W); t /= W;
    int y = (int)(t % H);
    int b = (int)(t / H);
    int a = c / D, d = c - a * D;

    float v = (float)src[idx];
    if (applyTanh) v = tanhf(v);
    long long p = (long long)priorOff + ((long long)y * W + x) * 3 + a;
    dst[((long long)b * Ptot + p) * D + d] = v;
}

// ---------------------------------------------------------------------------
// Priors: [19248, 4] = (cx, cy, w, h), square anchors, pixel scales / 550.
// ---------------------------------------------------------------------------
__global__ void priors_kernel(float* __restrict__ dst)
{
    const int bases[5]  = {0, 14283, 17958, 18930, 19173};
    const int sizes[5]  = {69, 35, 18, 9, 5};
    const float scales[5] = {24.f, 48.f, 96.f, 192.f, 384.f};
    int p = blockIdx.x * blockDim.x + threadIdx.x;
    if (p >= 19248) return;
    int l = 4;
    while (l > 0 && p < bases[l]) --l;
    int idx = p - bases[l];
    int cell = idx / 3;
    int a = idx - cell * 3;
    int s = sizes[l];
    int j = cell / s;
    int i = cell - j * s;
    float cx = (i + 0.5f) / (float)s;
    float cy = (j + 0.5f) / (float)s;
    float ar = (a == 0) ? 1.0f : (a == 1 ? 0.5f : 2.0f);
    float wh = scales[l] * sqrtf(ar) / 550.0f;
    dst[(long long)p * 4 + 0] = cx;
    dst[(long long)p * 4 + 1] = cy;
    dst[(long long)p * 4 + 2] = wh;
    dst[(long long)p * 4 + 3] = wh;
}

// ---------------------------------------------------------------------------
// Host-side pipeline
// ---------------------------------------------------------------------------
static void launch_conv(hipStream_t s, const _Float16* in, const _Float16* w,
                        const float* b, _Float16* out, int B, int Cin, int H, int W,
                        int Cout, int KH, int KW, int stride, int pad, int act)
{
    int OH = (H + 2 * pad - KH) / stride + 1;
    int OW = (W + 2 * pad - KW) / stride + 1;
    int mBlocks = (Cout + BM - 1) / BM;
    int nBlocks = (B * OH * OW + BN - 1) / BN;
    int ciShift = 31 - __builtin_clz((unsigned)Cin);   // Cin is a power of two
    conv_wmma_kernel<<<mBlocks * nBlocks, 256, 0, s>>>(
        in, w, b, out, B, Cin, H, W, Cout, KH, KW, stride, pad, OH, OW, act,
        ciShift, mBlocks, nBlocks);
}

static void launch_wcvt(hipStream_t s, const float* src, _Float16* dst,
                        int Cout, int Cin, int KHW)
{
    long long n = (long long)Cout * Cin * KHW;
    int blocks = (int)((n + 255) / 256);
    weight_rsc_f16_kernel<<<blocks, 256, 0, s>>>(src, dst, Cout, Cin, KHW);
}

static void launch_bilinear(hipStream_t s, const _Float16* src, _Float16* dst,
                            int B, int C, int Hs, int Ws, int Hd, int Wd, int accum)
{
    long long total = (long long)B * Hd * Wd * C;
    int blocks = (int)((total + 255) / 256);
    bilinear_kernel<<<blocks, 256, 0, s>>>(src, dst, B, C, Hs, Ws, Hd, Wd, accum);
}

extern "C" void kernel_launch(void* const* d_in, const int* in_sizes, int n_in,
                              void* d_out, int out_size, void* d_ws, size_t ws_size,
                              hipStream_t stream)
{
    (void)in_sizes; (void)n_in; (void)out_size; (void)ws_size;

    const int B = 8, F = 256;
    const int Ptot = 19248;

    const float* c3 = (const float*)d_in[0];
    const float* c4 = (const float*)d_in[1];
    const float* c5 = (const float*)d_in[2];
    const float* latw[3] = {(const float*)d_in[3], (const float*)d_in[5], (const float*)d_in[7]};
    const float* latb[3] = {(const float*)d_in[4], (const float*)d_in[6], (const float*)d_in[8]};
    const float* predw[3] = {(const float*)d_in[9],  (const float*)d_in[11], (const float*)d_in[13]};
    const float* predb[3] = {(const float*)d_in[10], (const float*)d_in[12], (const float*)d_in[14]};
    const float* downw[2] = {(const float*)d_in[15], (const float*)d_in[17]};
    const float* downb[2] = {(const float*)d_in[16], (const float*)d_in[18]};
    const float* protow[5] = {(const float*)d_in[19], (const float*)d_in[21], (const float*)d_in[23],
                              (const float*)d_in[25], (const float*)d_in[27]};
    const float* protob[5] = {(const float*)d_in[20], (const float*)d_in[22], (const float*)d_in[24],
                              (const float*)d_in[26], (const float*)d_in[28]};
    const float* upw = (const float*)d_in[29];  const float* upb = (const float*)d_in[30];
    const float* bbw = (const float*)d_in[31];  const float* bbb = (const float*)d_in[32];
    const float* cfw = (const float*)d_in[33];  const float* cfb = (const float*)d_in[34];
    const float* mkw = (const float*)d_in[35];  const float* mkb = (const float*)d_in[36];

    float* out = (float*)d_out;
    const long long BBOX_BASE  = 0;
    const long long CONF_BASE  = (long long)B * Ptot * 4;               // 615936
    const long long MASK_BASE  = CONF_BASE + (long long)B * Ptot * 81;  // 13088640
    const long long PRIOR_BASE = MASK_BASE + (long long)B * Ptot * 32;  // 18016128
    const long long PROTO_BASE = PRIOR_BASE + (long long)Ptot * 4;      // 18093120

    // Workspace: byte-carved, 256B-aligned chunks (16B alignment needed for v8h)
    char* wsb = (char*)d_ws;
    size_t o = 0;
    auto carve = [&](size_t elems) -> _Float16* {
        _Float16* p = (_Float16*)(wsb + o);
        o += ((elems * sizeof(_Float16)) + 255) & ~(size_t)255;
        return p;
    };

    const size_t SZ69  = (size_t)B * F * 69 * 69;
    const size_t SZ35  = (size_t)B * F * 35 * 35;
    const size_t SZ18  = (size_t)B * F * 18 * 18;
    const size_t SZ9   = (size_t)B * F * 9 * 9;
    const size_t SZ5   = (size_t)B * F * 5 * 5;
    const size_t SZ138 = (size_t)B * F * 138 * 138;

    _Float16* lat3 = carve(SZ69);
    _Float16* lat4 = carve(SZ35);
    _Float16* lat5 = carve(SZ18);
    _Float16* p3   = carve(SZ69);
    _Float16* p4   = carve(SZ35);
    _Float16* p5   = carve(SZ18);
    _Float16* p6   = carve(SZ9);
    _Float16* p7   = carve(SZ5);
    _Float16* U    = carve(SZ138);   // big tmp (proto 138^2, head upfeature)
    _Float16* V    = carve(SZ138);   // big tmp (proto 138^2, head conv outs)
    _Float16* pb   = carve(SZ69);    // proto ping buffer

    // NHWC f16 copies of inputs
    const size_t NC3 = (size_t)B * 512 * 69 * 69;
    const size_t NC4 = (size_t)B * 1024 * 35 * 35;
    const size_t NC5 = (size_t)B * 2048 * 18 * 18;
    _Float16* c3h = carve(NC3);
    _Float16* c4h = carve(NC4);
    _Float16* c5h = carve(NC5);

    // f16 weights re-laid-out to [oc][rs*Cin+ci]
    const size_t W33 = (size_t)F * F * 9;
    _Float16* latwh[3] = {carve((size_t)F * 512), carve((size_t)F * 1024), carve((size_t)F * 2048)};
    _Float16* predwh[3] = {carve(W33), carve(W33), carve(W33)};
    _Float16* downwh[2] = {carve(W33), carve(W33)};
    _Float16* protowh[5] = {carve(W33), carve(W33), carve(W33), carve(W33), carve((size_t)32 * F)};
    _Float16* upwh = carve(W33);
    _Float16* bbwh = carve((size_t)12 * F * 9);
    _Float16* cfwh = carve((size_t)243 * F * 9);
    _Float16* mkwh = carve((size_t)96 * F * 9);

    // ---------------- one-time conversions / re-layouts ----------------
    {
        int blocks;
        blocks = (int)((NC3 + 255) / 256);
        nchw_to_nhwc_f16_kernel<<<blocks, 256, 0, stream>>>(c3, c3h, B, 512, 69, 69);
        blocks = (int)((NC4 + 255) / 256);
        nchw_to_nhwc_f16_kernel<<<blocks, 256, 0, stream>>>(c4, c4h, B, 1024, 35, 35);
        blocks = (int)((NC5 + 255) / 256);
        nchw_to_nhwc_f16_kernel<<<blocks, 256, 0, stream>>>(c5, c5h, B, 2048, 18, 18);
    }
    launch_wcvt(stream, latw[0], latwh[0], F, 512, 1);
    launch_wcvt(stream, latw[1], latwh[1], F, 1024, 1);
    launch_wcvt(stream, latw[2], latwh[2], F, 2048, 1);
    for (int i = 0; i < 3; ++i) launch_wcvt(stream, predw[i], predwh[i], F, F, 9);
    for (int i = 0; i < 2; ++i) launch_wcvt(stream, downw[i], downwh[i], F, F, 9);
    for (int i = 0; i < 4; ++i) launch_wcvt(stream, protow[i], protowh[i], F, F, 9);
    launch_wcvt(stream, protow[4], protowh[4], 32, F, 1);
    launch_wcvt(stream, upw, upwh, F, F, 9);
    launch_wcvt(stream, bbw, bbwh, 12, F, 9);
    launch_wcvt(stream, cfw, cfwh, 243, F, 9);
    launch_wcvt(stream, mkw, mkwh, 96, F, 9);

    // ---------------- FPN ----------------
    launch_conv(stream, c5h, latwh[2], latb[2], lat5, B, 2048, 18, 18, F, 1, 1, 1, 0, 0);
    launch_conv(stream, c4h, latwh[1], latb[1], lat4, B, 1024, 35, 35, F, 1, 1, 1, 0, 0);
    launch_conv(stream, c3h, latwh[0], latb[0], lat3, B,  512, 69, 69, F, 1, 1, 1, 0, 0);
    launch_bilinear(stream, lat5, lat4, B, F, 18, 18, 35, 35, 1);  // m4 = lat4 + up(m5)
    launch_bilinear(stream, lat4, lat3, B, F, 35, 35, 69, 69, 1);  // m3 = lat3 + up(m4)
    launch_conv(stream, lat3, predwh[0], predb[0], p3, B, F, 69, 69, F, 3, 3, 1, 1, 1);
    launch_conv(stream, lat4, predwh[1], predb[1], p4, B, F, 35, 35, F, 3, 3, 1, 1, 1);
    launch_conv(stream, lat5, predwh[2], predb[2], p5, B, F, 18, 18, F, 3, 3, 1, 1, 1);
    launch_conv(stream, p5, downwh[0], downb[0], p6, B, F, 18, 18, F, 3, 3, 2, 1, 0);
    launch_conv(stream, p6, downwh[1], downb[1], p7, B, F,  9,  9, F, 3, 3, 2, 1, 0);

    // ---------------- Protonet (from P3) ----------------
    launch_conv(stream, p3,   protowh[0], protob[0], pb,   B, F, 69, 69, F, 3, 3, 1, 1, 1);
    launch_conv(stream, pb,   protowh[1], protob[1], lat3, B, F, 69, 69, F, 3, 3, 1, 1, 1);
    launch_conv(stream, lat3, protowh[2], protob[2], pb,   B, F, 69, 69, F, 3, 3, 1, 1, 1);
    launch_bilinear(stream, pb, U, B, F, 69, 69, 138, 138, 0);
    launch_conv(stream, U, protowh[3], protob[3], V, B, F, 138, 138, F, 3, 3, 1, 1, 1);
    launch_conv(stream, V, protowh[4], protob[4], lat3, B, F, 138, 138, 32, 1, 1, 1, 0, 1);
    {
        long long total = (long long)B * 138 * 138 * 32;   // already NHWC
        int blocks = (int)((total + 255) / 256);
        f16_to_f32_kernel<<<blocks, 256, 0, stream>>>(lat3, out + PROTO_BASE, total);
    }

    // ---------------- Prediction heads (shared weights) ----------------
    const _Float16* levels[5] = {p3, p4, p5, p6, p7};
    const int lh[5] = {69, 35, 18, 9, 5};
    const int poff[5] = {0, 14283, 17958, 18930, 19173};
    for (int l = 0; l < 5; ++l) {
        int H = lh[l], W = lh[l];
        launch_conv(stream, levels[l], upwh, upb, U, B, F, H, W, F, 3, 3, 1, 1, 1);
        long long hp_total;
        int blocks;
        // bbox: 3*4 channels
        launch_conv(stream, U, bbwh, bbb, V, B, F, H, W, 12, 3, 3, 1, 1, 0);
        hp_total = (long long)B * H * W * 12; blocks = (int)((hp_total + 255) / 256);
        head_permute_kernel<<<blocks, 256, 0, stream>>>(V, out + BBOX_BASE, B, H, W, 4, poff[l], Ptot, 0);
        // conf: 3*81 channels
        launch_conv(stream, U, cfwh, cfb, V, B, F, H, W, 243, 3, 3, 1, 1, 0);
        hp_total = (long long)B * H * W * 243; blocks = (int)((hp_total + 255) / 256);
        head_permute_kernel<<<blocks, 256, 0, stream>>>(V, out + CONF_BASE, B, H, W, 81, poff[l], Ptot, 0);
        // mask: 3*32 channels, tanh
        launch_conv(stream, U, mkwh, mkb, V, B, F, H, W, 96, 3, 3, 1, 1, 0);
        hp_total = (long long)B * H * W * 96; blocks = (int)((hp_total + 255) / 256);
        head_permute_kernel<<<blocks, 256, 0, stream>>>(V, out + MASK_BASE, B, H, W, 32, poff[l], Ptot, 1);
    }

    // ---------------- Priors ----------------
    priors_kernel<<<(Ptot + 255) / 256, 256, 0, stream>>>(out + PRIOR_BASE);
}
